// Model_13821204759268
// MI455X (gfx1250) — compile-verified, matrix operand order checked by reading, not run on hardware
//
#include <hip/hip_runtime.h>
#include <hip/hip_bf16.h>
#include <math.h>

// ---------------------------------------------------------------------------
// MI455X (gfx1250) LSTM-SRL forward.
// ~275 GFLOP total; weights L2-resident (192MB L2); HBM traffic ~1.1GB
// (~47us @ 23.3TB/s). Real cost: 4x256-step serial recurrence. Strategy:
//  - bulk GEMMs (x@W+b, h@Wout): 64x64-tile WMMA bf16 (f32 accum), direct
//    global b128 loads (verified: clause'd loads feeding v_wmma back-to-back)
//  - recurrence: persistent kernel, 32 WGs x 16 h-cols; U chunk (64KB bf16)
//    LDS-resident for the layer; h snapshot async-DMA'd to LDS each step
//    (GLOBAL_LOAD_ASYNC_TO_LDS_B128 / s_wait_asynccnt); h double-buffered in
//    L2 so only ONE device barrier per step.
// ---------------------------------------------------------------------------

#define B_   64
#define T_   256
#define CELL 512
#define EMB  511
#define GATE4 (4 * CELL)      // 2048
#define NLAB 64
#define NTOK (B_ * T_)        // 16384
#define NBLK_REC 32           // recurrent workgroups; 16 h-cols each

typedef __attribute__((ext_vector_type(16))) __bf16 v16bf;
typedef __attribute__((ext_vector_type(8)))  __bf16 v8bf;
typedef __attribute__((ext_vector_type(8)))  float  v8f;
typedef __attribute__((vector_size(16)))     int    v4i;

typedef __attribute__((address_space(1))) v4i* gptr_v4i;
typedef __attribute__((address_space(3))) v4i* lptr_v4i;

// ---- async 16B global->LDS copy (ASYNCcnt path), with sync fallback ------
__device__ __forceinline__ void async_copy16(void* lds, const void* g) {
#if __has_builtin(__builtin_amdgcn_global_load_async_to_lds_b128)
    __builtin_amdgcn_global_load_async_to_lds_b128(
        (gptr_v4i)(void*)g, (lptr_v4i)lds, 0, 0);
#else
    *(v8bf*)lds = *(const v8bf*)g;
#endif
}
__device__ __forceinline__ void async_wait0() {
#if __has_builtin(__builtin_amdgcn_global_load_async_to_lds_b128)
  #if __has_builtin(__builtin_amdgcn_s_wait_asynccnt)
    __builtin_amdgcn_s_wait_asynccnt(0);
  #else
    asm volatile("s_wait_asynccnt 0" ::: "memory");
  #endif
#endif
}

// ---- WMMA fragment load --------------------------------------------------
// 16-bit A/B operand layout (CDNA5 ISA 7.12.2): per-lane row/col = lane&15,
// K slots lanes 0-15 -> {k0..+7, k0+16..+23}, lanes 16-31 -> +8. Rows are
// K-contiguous so a fragment is two 16-byte loads.
__device__ __forceinline__ v16bf load_frag(const __bf16* __restrict__ row, int k0, int hi) {
    v8bf lo = *(const v8bf*)(row + k0 + hi);
    v8bf hh = *(const v8bf*)(row + k0 + 16 + hi);
    v16bf r;
#pragma unroll
    for (int i = 0; i < 8; ++i) { r[i] = lo[i]; r[i + 8] = hh[i]; }
    return r;
}

__device__ __forceinline__ v8f wmma_bf16(v16bf a, v16bf b, v8f c) {
    return __builtin_amdgcn_wmma_f32_16x16x32_bf16(false, a, false, b, (short)0, c, false, false);
}

// ---- device-wide barrier (agent scope, generation counter) ---------------
__device__ __forceinline__ void grid_barrier(unsigned* bar, unsigned nblk) {
    __syncthreads();
    if (threadIdx.x == 0) {
        unsigned* cnt = bar;
        unsigned* gen = bar + 1;
        unsigned g = __hip_atomic_load(gen, __ATOMIC_ACQUIRE, __HIP_MEMORY_SCOPE_AGENT);
        unsigned a = __hip_atomic_fetch_add(cnt, 1u, __ATOMIC_ACQ_REL, __HIP_MEMORY_SCOPE_AGENT);
        if (a + 1 == nblk) {
            __hip_atomic_store(cnt, 0u, __ATOMIC_RELAXED, __HIP_MEMORY_SCOPE_AGENT);
            __hip_atomic_store(gen, g + 1, __ATOMIC_RELEASE, __HIP_MEMORY_SCOPE_AGENT);
        } else {
            while (__hip_atomic_load(gen, __ATOMIC_ACQUIRE, __HIP_MEMORY_SCOPE_AGENT) == g)
                __builtin_amdgcn_s_sleep(8);
        }
    }
    __syncthreads();
}

// ---- transpose + f32->bf16 convert: src [mat][512][N] -> dst [mat][N][512]
__global__ void cvt_transpose_kernel(const float* __restrict__ src,
                                     __bf16* __restrict__ dst,
                                     int N, long total) {
    long idx = blockIdx.x * (long)blockDim.x + threadIdx.x;
    if (idx >= total) return;
    long k   = idx & 511;
    long mj  = idx >> 9;        // mat*N + j
    long j   = mj % N;
    long mat = mj / N;
    dst[idx] = (__bf16)src[(mat * 512 + k) * N + j];
}

// ---- embedding gather + predicate concat, store bf16 ---------------------
__global__ void embed_kernel(const int* __restrict__ wid, const int* __restrict__ pid,
                             const float* __restrict__ emb, __bf16* __restrict__ x0) {
    const int tok = blockIdx.x;                 // 0..16383
    const int w = wid[tok];
    const float p = (float)pid[tok];
    const float* e = emb + (size_t)w * EMB;
    for (int i = threadIdx.x; i < CELL; i += blockDim.x) {
        float v = (i < EMB) ? e[i] : p;
        x0[(size_t)tok * CELL + i] = (__bf16)v;
    }
}

// ---- bulk GEMM: C[M][ldc] = A[M][512](bf16) @ Bt[N][512]^T(bf16) + bias ---
__global__ void gemm_bf16_kernel(const __bf16* __restrict__ A,
                                 const __bf16* __restrict__ Bt,
                                 const float* __restrict__ bias,
                                 float* __restrict__ C, int ldc) {
    const int tid  = threadIdx.x;
    const int wave = tid >> 5, lane = tid & 31;
    const int row0 = blockIdx.x * 64;
    const int col0 = blockIdx.y * 64;
    const int mt   = wave & 3;
    const int nt0  = (wave >> 2) << 1;
    const int ri   = lane & 15;
    const int hi   = (lane & 16) ? 8 : 0;

    const __bf16* arow  = A  + (size_t)(row0 + mt * 16 + ri) * CELL;
    const __bf16* brow0 = Bt + (size_t)(col0 + nt0 * 16 + ri) * CELL;
    const __bf16* brow1 = brow0 + 16 * CELL;

    v8f acc0 = {}, acc1 = {};
#pragma unroll 4
    for (int k0 = 0; k0 < CELL; k0 += 32) {
        v16bf a  = load_frag(arow,  k0, hi);
        v16bf b0 = load_frag(brow0, k0, hi);
        v16bf b1 = load_frag(brow1, k0, hi);
        acc0 = wmma_bf16(a, b0, acc0);
        acc1 = wmma_bf16(a, b1, acc1);
    }
    const float bi0 = bias[col0 + nt0 * 16 + ri];
    const float bi1 = bias[col0 + (nt0 + 1) * 16 + ri];
#pragma unroll
    for (int r = 0; r < 8; ++r) {   // C layout: VGPR r -> row r(+8 hi lanes), col = lane&15
        int mm = row0 + mt * 16 + r + hi;
        C[(size_t)mm * ldc + col0 + nt0 * 16 + ri]       = acc0[r] + bi0;
        C[(size_t)mm * ldc + col0 + (nt0 + 1) * 16 + ri] = acc1[r] + bi1;
    }
}

// ---- persistent recurrent layer kernel -----------------------------------
// Per step: async-DMA snapshot of h (64KB) into LDS, z = h @ Uchunk via
// 256 WMMAs from LDS operands, f32 gates, masked carry, h written bf16 to
// the OTHER global h buffer (double-buffered) -> 1 device barrier per step.
__global__ void lstm_layer_kernel(const __bf16* __restrict__ Ut,   // [2048][512] this layer
                                  const float*  __restrict__ pre,  // [B*T][2048]
                                  const int*    __restrict__ mask, // [B][T]
                                  __bf16* __restrict__ hg0,        // [64][512] ping
                                  __bf16* __restrict__ hg1,        // [64][512] pong
                                  __bf16* __restrict__ xout,       // [B*T][512]
                                  int go_back,
                                  unsigned* __restrict__ bar) {
    extern __shared__ char smem[];
    __bf16* Ubuf = (__bf16*)smem;                    // [64][512] bf16 (64KB)
    __bf16* hlds = (__bf16*)(smem + 65536);          // [64][512] bf16 (64KB)
    float*  zbuf = (float*)(smem + 131072);          // [64][64]  (16KB)
    float*  cbuf = zbuf + 64 * 64;                   // [64][16] cell state
    float*  hbuf = cbuf + 64 * 16;                   // [64][16] hidden f32

    const int tid  = threadIdx.x;
    const int wave = tid >> 5, lane = tid & 31;
    const int cb   = blockIdx.x * 16;                // first owned h column

    // stage U chunk (async DMA): Ubuf row gn=g*16+n <- Ut row g*512+cb+n
    for (int c = tid; c < 4096; c += 256) {          // 4096 x 16B = 64KB
        int gn = c >> 6;                             // 64 chunks per 1KB row
        int ke = (c & 63) * 8;                       // element offset in row
        int g = gn >> 4, n = gn & 15;
        async_copy16(Ubuf + gn * 512 + ke,
                     Ut + (size_t)(g * CELL + cb + n) * CELL + ke);
    }
    for (int i = tid; i < 64 * 16; i += 256) {
        cbuf[i] = 0.f; hbuf[i] = 0.f;
        hg0[(i >> 4) * CELL + cb + (i & 15)] = (__bf16)0.f;
    }
    async_wait0();
    grid_barrier(bar, NBLK_REC);     // U staged, h ping buffer zeroed everywhere

    const int mt  = wave & 3;
    const int nt0 = (wave >> 2) << 1;
    const int ri  = lane & 15;
    const int hi  = (lane & 16) ? 8 : 0;
    const __bf16* arow  = hlds + (mt * 16 + ri) * 512;
    const __bf16* brow0 = Ubuf + (nt0 * 16 + ri) * 512;
    const __bf16* brow1 = brow0 + 16 * 512;

    for (int s = 0; s < T_; ++s) {
        const int j = go_back ? (T_ - 1 - s) : s;    // input & mask index
        const __bf16* hread = (s & 1) ? hg1 : hg0;
        __bf16*      hwrite = (s & 1) ? hg0 : hg1;

        // snapshot h -> LDS via async engine (64KB, 16B per lane-op)
        for (int c = tid; c < 4096; c += 256)
            async_copy16(hlds + c * 8, hread + c * 8);
        async_wait0();
        __syncthreads();

        // z(chunk) = h @ Uchunk : M=64, N=64, K=512, all operands in LDS
        v8f acc0 = {}, acc1 = {};
#pragma unroll 4
        for (int k0 = 0; k0 < CELL; k0 += 32) {
            v16bf a  = load_frag(arow,  k0, hi);
            v16bf b0 = load_frag(brow0, k0, hi);
            v16bf b1 = load_frag(brow1, k0, hi);
            acc0 = wmma_bf16(a, b0, acc0);
            acc1 = wmma_bf16(a, b1, acc1);
        }
#pragma unroll
        for (int r = 0; r < 8; ++r) {
            int mm = mt * 16 + r + hi;
            zbuf[mm * 64 + nt0 * 16 + ri]       = acc0[r];
            zbuf[mm * 64 + (nt0 + 1) * 16 + ri] = acc1[r];
        }
        __syncthreads();

        // gate math: 64 batch x 16 cols = 1024 elems, 4 per thread
#pragma unroll
        for (int q = 0; q < 4; ++q) {
            int e = tid + q * 256;
            int m = e >> 4, n = e & 15;
            const float* prow = pre + ((size_t)m * T_ + j) * GATE4 + cb + n;
            float zi = zbuf[m * 64 +  0 + n] + prow[0];
            float zf = zbuf[m * 64 + 16 + n] + prow[CELL];
            float zg = zbuf[m * 64 + 32 + n] + prow[2 * CELL];
            float zo = zbuf[m * 64 + 48 + n] + prow[3 * CELL];
            float ig = 1.f / (1.f + __expf(-zi));
            float fg = 1.f / (1.f + __expf(-zf));
            float gg = tanhf(zg);
            float og = 1.f / (1.f + __expf(-zo));
            float cn = fg * cbuf[e] + ig * gg;
            float hn = og * tanhf(cn);
            bool keep = mask[m * T_ + j] != 0;
            float c2 = keep ? cn : cbuf[e];
            float h2 = keep ? hn : hbuf[e];
            cbuf[e] = c2; hbuf[e] = h2;
            __bf16 hb = (__bf16)h2;
            hwrite[m * CELL + cb + n] = hb;
            xout[((size_t)m * T_ + s) * CELL + cb + n] = hb;
            if (s + 1 < T_) {       // prefetch next step's pre row into GL2
                const int jn = go_back ? (j - 1) : (j + 1);
                __builtin_prefetch(pre + ((size_t)m * T_ + jn) * GATE4 + cb + n, 0, 1);
            }
        }
        grid_barrier(bar, NBLK_REC);   // new h visible before next snapshot
    }
}

// ---------------------------------------------------------------------------
extern "C" void kernel_launch(void* const* d_in, const int* in_sizes, int n_in,
                              void* d_out, int out_size, void* d_ws, size_t ws_size,
                              hipStream_t stream) {
    const int*   wid  = (const int*)d_in[0];
    const int*   pid  = (const int*)d_in[1];
    const int*   mask = (const int*)d_in[2];
    const float* emb  = (const float*)d_in[3];
    const float* Ws   = (const float*)d_in[4];
    const float* Us   = (const float*)d_in[5];
    const float* bs   = (const float*)d_in[6];
    const float* Wout = (const float*)d_in[7];
    const float* bout = (const float*)d_in[8];
    float* out = (float*)d_out;

    char* ws = (char*)d_ws;
    size_t off = 0;
    float*  pre = (float*)(ws + off);   off += (size_t)NTOK * GATE4 * 4;   // 128 MB
    __bf16* x0  = (__bf16*)(ws + off);  off += (size_t)NTOK * CELL * 2;    // 16 MB
    __bf16* x1  = (__bf16*)(ws + off);  off += (size_t)NTOK * CELL * 2;    // 16 MB
    __bf16* Wt  = (__bf16*)(ws + off);  off += 4ull * GATE4 * CELL * 2;    // 8 MB
    __bf16* Utb = (__bf16*)(ws + off);  off += 4ull * GATE4 * CELL * 2;    // 8 MB
    __bf16* Wot = (__bf16*)(ws + off);  off += (size_t)NLAB * CELL * 2;    // 64 KB
    __bf16* hg0 = (__bf16*)(ws + off);  off += (size_t)B_ * CELL * 2;      // 64 KB
    __bf16* hg1 = (__bf16*)(ws + off);  off += (size_t)B_ * CELL * 2;      // 64 KB
    unsigned* bar = (unsigned*)(ws + off);

    (void)hipMemsetAsync(bar, 0, 2 * sizeof(unsigned), stream);

    {   // weight transpose+convert: Ws/Us [l][512][2048] -> [l][2048][512] bf16
        long tot = 4L * GATE4 * CELL;
        cvt_transpose_kernel<<<(tot + 255) / 256, 256, 0, stream>>>(Ws, Wt,  GATE4, tot);
        cvt_transpose_kernel<<<(tot + 255) / 256, 256, 0, stream>>>(Us, Utb, GATE4, tot);
        long tot2 = (long)NLAB * CELL;
        cvt_transpose_kernel<<<(tot2 + 255) / 256, 256, 0, stream>>>(Wout, Wot, NLAB, tot2);
    }
    embed_kernel<<<NTOK, 256, 0, stream>>>(wid, pid, emb, x0);

    __bf16* bufs[2] = { x0, x1 };
    const size_t shmem = 65536 + 65536 + 64 * 64 * 4 + 2 * 64 * 16 * 4;   // 152 KB
    for (int l = 0; l < 4; ++l) {
        __bf16* xin = bufs[l & 1];
        __bf16* xo  = bufs[(l + 1) & 1];
        gemm_bf16_kernel<<<dim3(NTOK / 64, GATE4 / 64), 256, 0, stream>>>(
            xin, Wt + (size_t)l * GATE4 * CELL, bs + (size_t)l * GATE4, pre, GATE4);
        lstm_layer_kernel<<<NBLK_REC, 256, shmem, stream>>>(
            Utb + (size_t)l * GATE4 * CELL, pre, mask, hg0, hg1, xo, l & 1, bar);
    }
    gemm_bf16_kernel<<<dim3(NTOK / 64, 1), 256, 0, stream>>>(bufs[0], Wot, bout, out, NLAB);
}
